// ScaledDotProductAttention_6631429505115
// MI455X (gfx1250) — matrix operand order repaired and checked by reading, not compiled, not run on hardware
//
#include <hip/hip_runtime.h>
#include <hip/hip_bf16.h>

#define B_ 16
#define T_ 2048
#define D_ 512
#define A_ 64

typedef __attribute__((ext_vector_type(16))) _Float16 v16h;
typedef __attribute__((ext_vector_type(8)))  _Float16 v8h;
typedef __attribute__((ext_vector_type(8)))  float    v8f;
typedef __attribute__((ext_vector_type(4)))  float    v4f;

__device__ inline v16h cat8(v8h lo, v8h hi) {
    v16h r;
#pragma unroll
    for (int i = 0; i < 8; i++) { r[i] = lo[i]; r[i + 8] = hi[i]; }
    return r;
}

// ---------------------------------------------------------------------------
// Kernel 1: weight prep. WqT/WkT = f16 transposed (A x D) so WMMA B-fragments
// are two contiguous 8-half runs each. wvsum[d] = sum_a Wv[d][a]; bvs = sum_a bv[a].
// ---------------------------------------------------------------------------
__global__ void prep_kernel(const float* __restrict__ Wq, const float* __restrict__ Wk,
                            const float* __restrict__ Wv, const float* __restrict__ bv,
                            _Float16* __restrict__ WqT, _Float16* __restrict__ WkT,
                            float* __restrict__ wvsum, float* __restrict__ bvs) {
    int tid = threadIdx.x;
    for (int i = tid; i < D_ * A_; i += 256) {
        int d = i / A_, a = i % A_;
        WqT[(size_t)a * D_ + d] = (_Float16)Wq[i];
        WkT[(size_t)a * D_ + d] = (_Float16)Wk[i];
    }
    for (int d = tid; d < D_; d += 256) {
        float s = 0.f;
        for (int a = 0; a < A_; a++) s += Wv[(size_t)d * A_ + a];
        wvsum[d] = s;
    }
    if (tid == 0) {
        float s = 0.f;
        for (int a = 0; a < A_; a++) s += bv[a];
        *bvs = s;
    }
}

// ---------------------------------------------------------------------------
// Kernel 2: Q/K projection via v_wmma_f32_16x16x32_f16. One wave per 16-token
// tile; 4 N-tiles (A=64) x 16 K-steps (D=512). The loaded inp floats are also
// dotted with wvsum to get vsum (V row-sums) for free. Epilogue: add bias,
// compute row sums -> q_mask/k_mask, store Qh/Kh f16, wvec = q_mask*vsum, kmvec.
// ---------------------------------------------------------------------------
__global__ __launch_bounds__(128) void proj_kernel(
    const float* __restrict__ inp, const float* __restrict__ bq, const float* __restrict__ bk,
    const _Float16* __restrict__ WqT, const _Float16* __restrict__ WkT,
    const float* __restrict__ wvsum, const float* __restrict__ bvs,
    _Float16* __restrict__ Qh, _Float16* __restrict__ Kh,
    float* __restrict__ wvec, float* __restrict__ kmvec) {
    int lane = threadIdx.x & 31;
    int tile = blockIdx.x * 4 + (threadIdx.x >> 5);   // 0..2047 over flat B*T/16
    long bt0 = (long)tile * 16;
    int n = lane & 15;     // A-frag row M / C-frag col N
    int g = lane >> 4;     // half-wave

    v8f zero = {};
    v8f accQ[4], accK[4];
#pragma unroll
    for (int j = 0; j < 4; j++) { accQ[j] = zero; accK[j] = zero; }
    float vpart = 0.f;

    const float* rowp = inp + (bt0 + n) * (size_t)D_;
    for (int k = 0; k < 16; k++) {
        int d0 = k * 32 + 8 * g;
        v4f f0 = *(const v4f*)(rowp + d0);
        v4f f1 = *(const v4f*)(rowp + d0 + 4);
        v4f f2 = *(const v4f*)(rowp + d0 + 16);
        v4f f3 = *(const v4f*)(rowp + d0 + 20);
        v4f w0 = *(const v4f*)(wvsum + d0);
        v4f w1 = *(const v4f*)(wvsum + d0 + 4);
        v4f w2 = *(const v4f*)(wvsum + d0 + 16);
        v4f w3 = *(const v4f*)(wvsum + d0 + 20);
        v16h a;
#pragma unroll
        for (int i = 0; i < 4; i++) {
            a[i]      = (_Float16)f0[i];
            a[4 + i]  = (_Float16)f1[i];
            a[8 + i]  = (_Float16)f2[i];
            a[12 + i] = (_Float16)f3[i];
            vpart += f0[i] * w0[i] + f1[i] * w1[i] + f2[i] * w2[i] + f3[i] * w3[i];
        }
#pragma unroll
        for (int j = 0; j < 4; j++) {
            const _Float16* wq = WqT + (size_t)(j * 16 + n) * D_ + k * 32 + 8 * g;
            v16h bfq = cat8(*(const v8h*)wq, *(const v8h*)(wq + 16));
            accQ[j] = __builtin_amdgcn_wmma_f32_16x16x32_f16(false, a, false, bfq,
                                                             (short)0, accQ[j], false, false);
            const _Float16* wk = WkT + (size_t)(j * 16 + n) * D_ + k * 32 + 8 * g;
            v16h bfk = cat8(*(const v8h*)wk, *(const v8h*)(wk + 16));
            accK[j] = __builtin_amdgcn_wmma_f32_16x16x32_f16(false, a, false, bfk,
                                                             (short)0, accK[j], false, false);
        }
    }

    // Epilogue: bias, row sums, f16 store
    float qs[8], ks[8];
#pragma unroll
    for (int r = 0; r < 8; r++) { qs[r] = 0.f; ks[r] = 0.f; }
#pragma unroll
    for (int j = 0; j < 4; j++) {
        float bqv = bq[j * 16 + n];
        float bkv = bk[j * 16 + n];
#pragma unroll
        for (int r = 0; r < 8; r++) {
            accQ[j][r] += bqv; accK[j][r] += bkv;
            qs[r] += accQ[j][r]; ks[r] += accK[j][r];
        }
    }
#pragma unroll
    for (int j = 0; j < 4; j++) {
#pragma unroll
        for (int r = 0; r < 8; r++) {
            size_t idx = (size_t)(bt0 + 8 * g + r) * A_ + j * 16 + n;
            Qh[idx] = (_Float16)accQ[j][r];
            Kh[idx] = (_Float16)accK[j][r];
        }
    }
    // reduce row sums over the 16 lanes (cols) of each half-wave
#pragma unroll
    for (int mk = 1; mk < 16; mk <<= 1) {
#pragma unroll
        for (int r = 0; r < 8; r++) {
            qs[r] += __shfl_xor(qs[r], mk, 32);
            ks[r] += __shfl_xor(ks[r], mk, 32);
        }
    }
    float vall = vpart + __shfl_xor(vpart, 16, 32);  // full D dot for row n
    float bvsv = *bvs;
#pragma unroll
    for (int r = 0; r < 8; r++) {
        float vr = __shfl(vall, 8 * g + r, 32);      // vsum for row 8g+r
        if (n == 0) {                                 // lanes 0 and 16
            float qm = (qs[r] != 0.f) ? 1.f : 0.f;    // sign(|sum Q|)
            float km = (ks[r] != 0.f) ? 1.f : 0.f;    // sign(|sum K|)
            wvec[bt0 + 8 * g + r]  = qm * (vr + bvsv);
            kmvec[bt0 + 8 * g + r] = km;
        }
    }
}

// ---------------------------------------------------------------------------
// Kernel 3: flash-style attention. One wave per 16-query tile; streams all 128
// key tiles with 2 WMMAs each (A=64), per-lane online softmax over owned
// columns, final 16-lane log-sum-exp merge. Emits c[b,t] = sum_s att*w[s].
// ---------------------------------------------------------------------------
__global__ __launch_bounds__(128) void attn_kernel(
    const _Float16* __restrict__ Qh, const _Float16* __restrict__ Kh,
    const float* __restrict__ wvec, const float* __restrict__ kmvec,
    float* __restrict__ cvec) {
    int lane = threadIdx.x & 31;
    int tile = blockIdx.x * 4 + (threadIdx.x >> 5);   // 0..2047
    int b  = tile >> 7;
    int t0 = (tile & 127) << 4;
    int n = lane & 15, g = lane >> 4;

    size_t qbase = ((size_t)b * T_ + t0 + n) * A_;
    v16h QA0 = cat8(*(const v8h*)(Qh + qbase + 8 * g),      *(const v8h*)(Qh + qbase + 16 + 8 * g));
    v16h QA1 = cat8(*(const v8h*)(Qh + qbase + 32 + 8 * g), *(const v8h*)(Qh + qbase + 48 + 8 * g));

    float mR[8], lR[8], cR[8];
#pragma unroll
    for (int r = 0; r < 8; r++) { mR[r] = -3.0e38f; lR[r] = 0.f; cR[r] = 0.f; }

    const float* wv  = wvec  + (size_t)b * T_;
    const float* kmv = kmvec + (size_t)b * T_;
    const _Float16* Kb = Kh + (size_t)b * T_ * A_;

    for (int s0 = 0; s0 < T_; s0 += 16) {
        const _Float16* kr = Kb + (size_t)(s0 + n) * A_;
        v16h KB0 = cat8(*(const v8h*)(kr + 8 * g),      *(const v8h*)(kr + 16 + 8 * g));
        v16h KB1 = cat8(*(const v8h*)(kr + 32 + 8 * g), *(const v8h*)(kr + 48 + 8 * g));
        v8f acc = {};
        acc = __builtin_amdgcn_wmma_f32_16x16x32_f16(false, QA0, false, KB0, (short)0, acc, false, false);
        acc = __builtin_amdgcn_wmma_f32_16x16x32_f16(false, QA1, false, KB1, (short)0, acc, false, false);
        float w  = wv[s0 + n];
        float km = kmv[s0 + n];
#pragma unroll
        for (int r = 0; r < 8; r++) {
            float x = acc[r] * 0.125f;            // / sqrt(A)
            x = (km == 0.f) ? 1e-8f : x;          // K padding mask (pre-softmax)
            float mo = mR[r];
            float mn = fmaxf(mo, x);
            float sc = __expf(mo - mn);
            float p  = __expf(x - mn);
            lR[r] = lR[r] * sc + p;
            cR[r] = cR[r] * sc + p * w;
            mR[r] = mn;
        }
    }
    // merge the 16 per-lane online-softmax states of each half-wave
#pragma unroll
    for (int mk = 1; mk < 16; mk <<= 1) {
#pragma unroll
        for (int r = 0; r < 8; r++) {
            float om = __shfl_xor(mR[r], mk, 32);
            float ol = __shfl_xor(lR[r], mk, 32);
            float oc = __shfl_xor(cR[r], mk, 32);
            float mn = fmaxf(mR[r], om);
            float s1 = __expf(mR[r] - mn);
            float s2 = __expf(om - mn);
            lR[r] = lR[r] * s1 + ol * s2;
            cR[r] = cR[r] * s1 + oc * s2;
            mR[r] = mn;
        }
    }
    if (n == 0) {
#pragma unroll
        for (int r = 0; r < 8; r++)
            cvec[(size_t)b * T_ + t0 + 8 * g + r] = cR[r] / lR[r];
    }
}

// ---------------------------------------------------------------------------
// Kernels 4/5: out[b,d] = sum_t inp[b,t,d] * c[b,t], deterministic two-stage.
// ---------------------------------------------------------------------------
__global__ __launch_bounds__(256) void finpart_kernel(const float* __restrict__ inp,
                                                      const float* __restrict__ cvec,
                                                      float* __restrict__ part) {
    int b = blockIdx.x, dc = blockIdx.y, ts = blockIdx.z;
    int d = dc * 256 + threadIdx.x;
    const float* ip = inp + ((size_t)b * T_ + ts * 128) * D_ + d;
    const float* cp = cvec + (size_t)b * T_ + ts * 128;
    float acc = 0.f;
    for (int t = 0; t < 128; t++) acc += ip[(size_t)t * D_] * cp[t];
    part[((size_t)b * D_ + d) * 16 + ts] = acc;
}

__global__ __launch_bounds__(256) void finred_kernel(const float* __restrict__ part,
                                                     float* __restrict__ out) {
    int i = blockIdx.x * 256 + threadIdx.x;   // 0..8191 = b*D + d
    float s = 0.f;
#pragma unroll
    for (int ts = 0; ts < 16; ts++) s += part[(size_t)i * 16 + ts];
    out[i] = s;
}

extern "C" void kernel_launch(void* const* d_in, const int* in_sizes, int n_in,
                              void* d_out, int out_size, void* d_ws, size_t ws_size,
                              hipStream_t stream) {
    const float* inp = (const float*)d_in[0];
    const float* Wq  = (const float*)d_in[1];
    const float* bq  = (const float*)d_in[2];
    const float* Wk  = (const float*)d_in[3];
    const float* bk  = (const float*)d_in[4];
    const float* Wv  = (const float*)d_in[5];
    const float* bv  = (const float*)d_in[6];

    char* ws = (char*)d_ws;
    _Float16* Qh    = (_Float16*)(ws + 0);          // 4 MB  (B*T*A f16)
    _Float16* Kh    = (_Float16*)(ws + 4194304);    // 4 MB
    _Float16* WqT   = (_Float16*)(ws + 8388608);    // 64 KB (A*D f16)
    _Float16* WkT   = (_Float16*)(ws + 8454144);    // 64 KB
    float*    wvsum = (float*)   (ws + 8519680);    // 2 KB
    float*    bvs   = (float*)   (ws + 8521728);    // 256 B
    float*    wvec  = (float*)   (ws + 8521984);    // 128 KB (B*T)
    float*    kmvec = (float*)   (ws + 8653056);    // 128 KB
    float*    cvec  = (float*)   (ws + 8784128);    // 128 KB
    float*    part  = (float*)   (ws + 8915200);    // 512 KB (B*D*16)

    prep_kernel<<<1, 256, 0, stream>>>(Wq, Wk, Wv, bv, WqT, WkT, wvsum, bvs);
    proj_kernel<<<512, 128, 0, stream>>>(inp, bq, bk, WqT, WkT, wvsum, bvs,
                                         Qh, Kh, wvec, kmvec);
    attn_kernel<<<512, 128, 0, stream>>>(Qh, Kh, wvec, kmvec, cvec);
    finpart_kernel<<<dim3(B_, 2, 16), 256, 0, stream>>>(inp, cvec, part);
    finred_kernel<<<32, 256, 0, stream>>>(part, (float*)d_out);
}
